// APNET_7627861918083
// MI455X (gfx1250) — matrix operand order repaired and checked by reading, not compiled, not run on hardware
//
#include <hip/hip_runtime.h>
#include <hip/hip_bf16.h>

typedef __attribute__((ext_vector_type(16))) _Float16 v16h;
typedef __attribute__((ext_vector_type(8)))  float    v8f;

#define N_TOK 49
#define NP    64
#define DIM   256
#define HEADS 8
#define DH    32
#define NWIN  4096
#define SCALE 0.1767766952966369f   // 1/sqrt(32)

// LDS leading dims (f16 elements), chosen 16B-aligned rows + bank-conflict-free
#define XH_LD 264   // 528 B/row -> 132 dw -> +4 banks/row
#define QK_LD 40    // 80 B/row  -> 20 dw  -> distinct banks over 16 rows
#define VT_LD 72    // 144 B/row -> 36 dw  -> distinct banks over 16 rows
#define AT_LD 72

// ---- WMMA fragment helpers (ISA 7.12.2 layouts, wave32) -------------------

// A fragment (16x32 f16) from row-major src[row][k], ld in elements.
// lane: M = lane%16, half = lane/16; elems e<8 -> K=half*8+e, e>=8 -> K=16+half*8+(e-8)
__device__ __forceinline__ v16h load_frag_A(const _Float16* src, int ld,
                                            int row_base, int k_base) {
  const int lane = threadIdx.x & 31;
  const int ln = lane & 15, hg = lane >> 4;
  const _Float16* p = src + (size_t)(row_base + ln) * ld + k_base + hg * 8;
  union { v16h v; float4 q[2]; } u;
  u.q[0] = *(const float4*)(p);        // K = half*8 + 0..7
  u.q[1] = *(const float4*)(p + 16);   // K = 16 + half*8 + 0..7
  return u.v;
}

// B fragment (32x16 f16) from TRANSPOSED row-major src[n][k], ld in elements.
// lane: N = lane%16, half = lane/16; elems e<8 -> K=half*8+e, e>=8 -> K=16+half*8+(e-8)
__device__ __forceinline__ v16h load_frag_B(const _Float16* src, int ld,
                                            int k_base, int col_base) {
  const int lane = threadIdx.x & 31;
  const int ln = lane & 15, hg = lane >> 4;
  const _Float16* p = src + (size_t)(col_base + ln) * ld + k_base + hg * 8;
  union { v16h v; float4 q[2]; } u;
  u.q[0] = *(const float4*)(p);        // K = half*8 + 0..7
  u.q[1] = *(const float4*)(p + 16);   // K = 16 + half*8 + 0..7
  return u.v;
}

__device__ __forceinline__ v8f wmma_f16(v16h a, v16h b, v8f c) {
  return __builtin_amdgcn_wmma_f32_16x16x32_f16(false, a, false, b,
                                                (short)0, c, false, false);
}

// Async DMA one 16-byte chunk per lane from global into LDS (ASYNCcnt path).
__device__ __forceinline__ void async_copy_b128(unsigned lds_byte_off,
                                                const void* gaddr) {
  asm volatile("global_load_async_to_lds_b128 %0, %1, off"
               :: "v"(lds_byte_off),
                  "v"((unsigned long long)(uintptr_t)gaddr)
               : "memory");
}
__device__ __forceinline__ void wait_async0() {
  asm volatile("s_wait_asynccnt 0x0" ::: "memory");
}

// ---- Kernel 0: weight transpose+convert, bias gather ----------------------

__global__ __launch_bounds__(256) void prep_kernel(
    const float* __restrict__ W_qkv, const float* __restrict__ W_out,
    const float* __restrict__ bias_table, const int* __restrict__ rel_idx,
    _Float16* __restrict__ WqkvT, _Float16* __restrict__ WoutT,
    float* __restrict__ biasH) {
  int id = blockIdx.x * 256 + threadIdx.x;
  if (id < 256 * 768) {
    int j = id >> 8, i = id & 255;                    // j: out col (768), i: k (256)
    WqkvT[j * 256 + i] = (_Float16)W_qkv[i * 768 + j];
  } else if (id < 256 * 768 + 256 * 256) {
    int k = id - 256 * 768;
    int j = k >> 8, i = k & 255;
    WoutT[j * 256 + i] = (_Float16)W_out[i * 256 + j];
  } else if (id < 256 * 768 + 256 * 256 + HEADS * NP * NP) {
    int k = id - (256 * 768 + 256 * 256);
    int h = k >> 12, rem = k & 4095;
    int i = rem >> 6, j = rem & 63;
    float v = -1e30f;                                 // masks padded cols in softmax
    if (i < N_TOK && j < N_TOK)
      v = bias_table[rel_idx[i * N_TOK + j] * HEADS + h];
    biasH[h * NP * NP + i * NP + j] = v;
  }
}

// ---- Kernel 1: fused QKV + attention per window ---------------------------

__global__ __launch_bounds__(256) void attn_win_kernel(
    const float* __restrict__ x, const _Float16* __restrict__ WqkvT,
    const float* __restrict__ biasH, _Float16* __restrict__ Oh) {
  __shared__ _Float16 Xh[NP * XH_LD];        // 33792 B: x window, f16, padded rows 0
  __shared__ _Float16 Qs[NP * QK_LD];        //  5120 B: Q (pre-scaled), row=token
  __shared__ _Float16 Ks[NP * QK_LD];        //  5120 B: K, row=token (B-op for Q@K^T)
  __shared__ _Float16 Vt[DH * VT_LD];        //  4608 B: V transposed (B-op for A@V)
  __shared__ _Float16 At[NP * AT_LD];        //  9216 B: attn probs f16

  const int win  = blockIdx.x;
  const int tid  = threadIdx.x;
  const int wave = tid >> 5;
  const int lane = tid & 31;
  const int ln   = lane & 15;
  const int hg   = lane >> 4;

  // Hint the weight slab toward the caches (shared by all 4096 blocks).
  __builtin_prefetch(WqkvT + (size_t)tid * 256, 0, 1);

  // Load x window -> f16 LDS; zero padded rows (avoid NaN via 0*garbage in WMMA)
  const float* xp = x + (size_t)win * N_TOK * DIM;
  for (int idx = tid; idx < NP * DIM; idx += 256) {
    int r = idx >> 8, c = idx & 255;
    float v = (r < N_TOK) ? xp[r * DIM + c] : 0.0f;
    Xh[r * XH_LD + c] = (_Float16)v;
  }
  __syncthreads();

  _Float16* ohp = Oh + (size_t)win * NP * DIM;

  for (int h = 0; h < HEADS; ++h) {
    // ---- QKV for this head: 24 C-tiles (Q/K/V x 4 row-tiles x 2 col-tiles)
    #pragma unroll
    for (int i = 0; i < 3; ++i) {
      int job = wave * 3 + i;
      int mat = job >> 3;                 // 0=Q 1=K 2=V
      int rt  = (job & 7) >> 1;
      int ct  = job & 1;
      const _Float16* wB =
          WqkvT + (size_t)(mat * DIM + h * DH + ct * 16) * DIM;
      v8f acc = {};
      #pragma unroll
      for (int ks = 0; ks < 8; ++ks) {
        v16h a = load_frag_A(Xh, XH_LD, rt * 16, ks * 32);
        v16h b = load_frag_B(wB, DIM, ks * 32, 0);
        acc = wmma_f16(a, b, acc);
      }
      if (mat == 0) {
        #pragma unroll
        for (int r = 0; r < 8; ++r)
          Qs[(rt * 16 + hg * 8 + r) * QK_LD + ct * 16 + ln] =
              (_Float16)(acc[r] * SCALE);
      } else if (mat == 1) {
        #pragma unroll
        for (int r = 0; r < 8; ++r)
          Ks[(rt * 16 + hg * 8 + r) * QK_LD + ct * 16 + ln] = (_Float16)acc[r];
      } else {
        #pragma unroll
        for (int r = 0; r < 8; ++r)      // store transposed: Vt[dh][token]
          Vt[(ct * 16 + ln) * VT_LD + rt * 16 + hg * 8 + r] = (_Float16)acc[r];
      }
    }
    __syncthreads();

    // ---- sim = Q @ K^T (+bias), in-register softmax. Waves 0..3, one row-tile each.
    if (wave < 4) {
      int rt = wave;
      v16h aq = load_frag_A(Qs, QK_LD, rt * 16, 0);   // K-dim = 32 = one WMMA step
      v8f s[4];
      #pragma unroll
      for (int ct = 0; ct < 4; ++ct) {
        v16h bk = load_frag_B(Ks, QK_LD, 0, ct * 16);
        v8f z = {};
        s[ct] = wmma_f16(aq, bk, z);
      }
      const float* bp = biasH + h * NP * NP;
      #pragma unroll
      for (int ct = 0; ct < 4; ++ct)
        #pragma unroll
        for (int r = 0; r < 8; ++r)
          s[ct][r] += bp[(rt * 16 + hg * 8 + r) * NP + ct * 16 + ln];

      // C layout: row (rt*16+hg*8+r) lives entirely in one 16-lane half-group
      #pragma unroll
      for (int r = 0; r < 8; ++r) {
        float m = fmaxf(fmaxf(s[0][r], s[1][r]), fmaxf(s[2][r], s[3][r]));
        #pragma unroll
        for (int off = 8; off >= 1; off >>= 1)
          m = fmaxf(m, __shfl_xor(m, off, 16));
        float e0 = __expf(s[0][r] - m), e1 = __expf(s[1][r] - m);
        float e2 = __expf(s[2][r] - m), e3 = __expf(s[3][r] - m);
        float sum = e0 + e1 + e2 + e3;
        #pragma unroll
        for (int off = 8; off >= 1; off >>= 1)
          sum += __shfl_xor(sum, off, 16);
        float inv = 1.0f / sum;                       // sum >= 1 always
        int row = rt * 16 + hg * 8 + r;
        At[row * AT_LD + 0 * 16 + ln] = (_Float16)(e0 * inv);
        At[row * AT_LD + 1 * 16 + ln] = (_Float16)(e1 * inv);
        At[row * AT_LD + 2 * 16 + ln] = (_Float16)(e2 * inv);
        At[row * AT_LD + 3 * 16 + ln] = (_Float16)(e3 * inv);
      }
    }
    __syncthreads();

    // ---- out_h = attn @ V : 8 C-tiles (4 row-tiles x 2 dh-tiles), one per wave
    {
      int rt = wave >> 1, ct = wave & 1;
      v8f acc = {};
      #pragma unroll
      for (int ks = 0; ks < 2; ++ks) {
        v16h a = load_frag_A(At, AT_LD, rt * 16, ks * 32);
        v16h b = load_frag_B(Vt, VT_LD, ks * 32, ct * 16);
        acc = wmma_f16(a, b, acc);
      }
      #pragma unroll
      for (int r = 0; r < 8; ++r)
        ohp[(size_t)(rt * 16 + hg * 8 + r) * DIM + h * DH + ct * 16 + ln] =
            (_Float16)acc[r];
    }
    __syncthreads();
  }
}

// ---- Kernel 2: output projection, per-window 64x256 @ 256x256 -------------
// A-tile (Oh window) staged into LDS via the gfx1250 async-DMA path.

__global__ __launch_bounds__(256) void out_proj_kernel(
    const _Float16* __restrict__ Oh, const _Float16* __restrict__ WoutT,
    float* __restrict__ out) {
  __shared__ _Float16 Ohs[NP * XH_LD];       // 33792 B, padded rows

  const int win  = blockIdx.x;
  const int wave = threadIdx.x >> 5;
  const int lane = threadIdx.x & 31;
  const int ln   = lane & 15;
  const int hg   = lane >> 4;

  const _Float16* A = Oh + (size_t)win * NP * DIM;

  // Stage 64x256 f16 (32 KB) into LDS: one row (512 B) per wave-op,
  // 16 B per lane, 8 async ops per wave. Tracked by ASYNCcnt.
  for (int row = wave; row < NP; row += 8) {
    unsigned lds_off =
        (unsigned)(uintptr_t)(&Ohs[row * XH_LD]) + (unsigned)(lane * 16);
    async_copy_b128(lds_off, (const char*)(A + (size_t)row * DIM) + lane * 16);
  }
  wait_async0();
  __syncthreads();

  const int rt = wave >> 1;          // row tile 0..3
  const int cb = (wave & 1) * 8;     // 8 col tiles per wave -> 16 tiles/row-tile

  v8f acc[8] = {};
  for (int ks = 0; ks < 8; ++ks) {
    v16h a = load_frag_A(Ohs, XH_LD, rt * 16, ks * 32);
    #pragma unroll
    for (int c = 0; c < 8; ++c) {
      v16h b = load_frag_B(WoutT, DIM, ks * 32, (cb + c) * 16);
      acc[c] = wmma_f16(a, b, acc[c]);
    }
  }

  float* op = out + (size_t)win * N_TOK * DIM;
  #pragma unroll
  for (int c = 0; c < 8; ++c)
    #pragma unroll
    for (int r = 0; r < 8; ++r) {
      int row = rt * 16 + hg * 8 + r;
      if (row < N_TOK)
        op[(size_t)row * DIM + (cb + c) * 16 + ln] = acc[c][r];
    }
}

// ---- Launch ---------------------------------------------------------------

extern "C" void kernel_launch(void* const* d_in, const int* in_sizes, int n_in,
                              void* d_out, int out_size, void* d_ws,
                              size_t ws_size, hipStream_t stream) {
  (void)in_sizes; (void)n_in; (void)out_size; (void)ws_size;
  const float* x          = (const float*)d_in[0];
  const float* W_qkv      = (const float*)d_in[1];
  const float* W_out      = (const float*)d_in[2];
  const float* bias_table = (const float*)d_in[3];
  const int*   rel_idx    = (const int*)d_in[4];

  char* ws = (char*)d_ws;
  _Float16* WqkvT = (_Float16*)(ws);                 // 768x256 f16 = 393216 B
  _Float16* WoutT = (_Float16*)(ws + 393216);        // 256x256 f16 = 131072 B
  float*    biasH = (float*)   (ws + 524288);        // 8x64x64 f32 = 131072 B
  _Float16* Oh    = (_Float16*)(ws + 655360);        // 4096x64x256 f16

  prep_kernel<<<1152, 256, 0, stream>>>(W_qkv, W_out, bias_table, rel_idx,
                                        WqkvT, WoutT, biasH);
  attn_win_kernel<<<NWIN, 256, 0, stream>>>(x, WqkvT, biasH, Oh);
  out_proj_kernel<<<NWIN, 256, 0, stream>>>(Oh, WoutT, (float*)d_out);
}
